// GQAAttention_84928683311387
// MI455X (gfx1250) — matrix operand order repaired and checked by reading, not compiled, not run on hardware
//
#include <hip/hip_runtime.h>
#include <hip/hip_bf16.h>
#include <math.h>

#define S_LEN 2048
#define HID   4096
#define NH    32
#define NKV   8
#define HD    128
#define GQ    (NH / NKV)      // 4
#define QDIM  (NH * HD)       // 4096
#define KVDIM (NKV * HD)      // 1024

typedef __bf16 v16bf __attribute__((ext_vector_type(16)));
typedef float  v8f   __attribute__((ext_vector_type(8)));

union Frag { v16bf v; uint4 q[2]; };

__device__ __forceinline__ unsigned short f2bf(float f) {
  union { float f; unsigned u; } x; x.f = f;
  unsigned r = x.u + 0x7fffu + ((x.u >> 16) & 1u);   // round-to-nearest-even
  return (unsigned short)(r >> 16);
}
__device__ __forceinline__ float bf2f(unsigned short h) {
  union { unsigned u; float f; } x; x.u = ((unsigned)h) << 16;
  return x.f;
}

// ---------------------------------------------------------------------------
// fp32 -> bf16 convert (vectorized)
// ---------------------------------------------------------------------------
__global__ __launch_bounds__(256) void cvt_f32_bf16(
    const float* __restrict__ src, unsigned short* __restrict__ dst, int n4) {
  int i = blockIdx.x * blockDim.x + threadIdx.x;
  if (i >= n4) return;
  float4 f = reinterpret_cast<const float4*>(src)[i];
  ushort4 o;
  o.x = f2bf(f.x); o.y = f2bf(f.y); o.z = f2bf(f.z); o.w = f2bf(f.w);
  reinterpret_cast<ushort4*>(dst)[i] = o;
}

// ---------------------------------------------------------------------------
// W fp32 [K,N] -> Wt bf16 [N,K]  (LDS tile transpose, coalesced both sides)
// ---------------------------------------------------------------------------
__global__ __launch_bounds__(256) void transpose_cvt(
    const float* __restrict__ W, unsigned short* __restrict__ Wt, int K, int N) {
  __shared__ float tile[32][33];
  const int tx = threadIdx.x & 31;
  const int ty = threadIdx.x >> 5;       // 0..7
  const int bn = blockIdx.x * 32;        // along N
  const int bk = blockIdx.y * 32;        // along K
#pragma unroll
  for (int j = 0; j < 32; j += 8)
    tile[ty + j][tx] = W[(size_t)(bk + ty + j) * N + bn + tx];
  __syncthreads();
#pragma unroll
  for (int j = 0; j < 32; j += 8)
    Wt[(size_t)(bn + ty + j) * K + bk + tx] = f2bf(tile[tx][ty + j]);
}

// ---------------------------------------------------------------------------
// GEMM: C[M,N] = A[M,K](bf16) * Bt[N,K](bf16)^T + bias
// Block: 128 threads (4 waves) -> 128x64 C tile; wave w owns rows 32w..32w+31
// (two 16-row WMMA strips), all 64 columns (4 n-tiles) -> 8 WMMA per k-step.
// B tile (64 cols x 32 k, shared by all 4 waves) is double-buffered in LDS and
// filled with GLOBAL_LOAD_ASYNC_TO_LDS_B128 (ASYNCcnt pipelined): each thread
// issues exactly 2 async b128s per stage, so `s_wait_asynccnt 2` retires the
// previous stage while the next stays in flight.
// A fragments are direct global b128 loads (rows private to the wave,
// contiguous, L2-resident).
// ---------------------------------------------------------------------------
#define BROW 40   // padded LDS row stride (elements) to spread banks

template <bool F32OUT>
__global__ __launch_bounds__(128) void gemm_bf16_wmma(
    const unsigned short* __restrict__ A,
    const unsigned short* __restrict__ Bt,
    const float* __restrict__ bias,
    void* __restrict__ Cout, int M, int N, int K) {
  __shared__ unsigned short Blds[2][64 * BROW];

  const int tid  = threadIdx.x;
  const int wave = tid >> 5;
  const int lane = tid & 31;
  const int lr = lane & 15, lh = lane >> 4;
  const int row0 = blockIdx.y * 128 + wave * 32;
  const int col0 = blockIdx.x * 64;

  // async-copy mapping: 2 threads per column, 32 bytes (2 x b128) each
  const int bcol  = tid >> 1;          // 0..63
  const int bhalf = (tid & 1) * 16;    // element offset in the 32-wide k slice

  v8f acc[2][4];
#pragma unroll
  for (int mi = 0; mi < 2; ++mi)
#pragma unroll
    for (int nb = 0; nb < 4; ++nb)
#pragma unroll
      for (int j = 0; j < 8; ++j) acc[mi][nb][j] = 0.f;

  const unsigned short* arow0 = A + (size_t)(row0 + lr) * K;
  const unsigned short* arow1 = A + (size_t)(row0 + 16 + lr) * K;
  const unsigned short* bsrc  = Bt + (size_t)(col0 + bcol) * K + bhalf;

  auto stage = [&](int buf, int k0) {
    const unsigned short* src = bsrc + k0;
    unsigned lds = (unsigned)(size_t)&Blds[buf][bcol * BROW + bhalf];
    asm volatile("global_load_async_to_lds_b128 %0, %1, off"
                 :: "v"(lds), "v"(src) : "memory");
    asm volatile("global_load_async_to_lds_b128 %0, %1, off offset:16"
                 :: "v"(lds), "v"(src) : "memory");
  };

  stage(0, 0);
  int buf = 0;
  for (int k0 = 0; k0 < K; k0 += 32) {
    const bool has_next = (k0 + 32) < K;
    if (has_next) {
      stage(buf ^ 1, k0 + 32);
      asm volatile("s_wait_asynccnt 0x2" ::: "memory");
    } else {
      asm volatile("s_wait_asynccnt 0x0" ::: "memory");
    }
    __syncthreads();   // LDS B tile for this k-step visible to all waves

    Frag a0, a1;
    const unsigned short* ap0 = arow0 + k0 + lh * 8;
    const unsigned short* ap1 = arow1 + k0 + lh * 8;
    a0.q[0] = *reinterpret_cast<const uint4*>(ap0);
    a0.q[1] = *reinterpret_cast<const uint4*>(ap0 + 16);
    a1.q[0] = *reinterpret_cast<const uint4*>(ap1);
    a1.q[1] = *reinterpret_cast<const uint4*>(ap1 + 16);

#pragma unroll
    for (int nb = 0; nb < 4; ++nb) {
      Frag b;
      const unsigned short* bp = &Blds[buf][(nb * 16 + lr) * BROW + lh * 8];
      b.q[0] = *reinterpret_cast<const uint4*>(bp);
      b.q[1] = *reinterpret_cast<const uint4*>(bp + 16);
      acc[0][nb] = __builtin_amdgcn_wmma_f32_16x16x32_bf16(
          false, a0.v, false, b.v, (short)0, acc[0][nb], false, false);
      acc[1][nb] = __builtin_amdgcn_wmma_f32_16x16x32_bf16(
          false, a1.v, false, b.v, (short)0, acc[1][nb], false, false);
    }
    __syncthreads();   // all reads done before next stage overwrites buf
    buf ^= 1;
  }

#pragma unroll
  for (int mi = 0; mi < 2; ++mi)
#pragma unroll
    for (int nb = 0; nb < 4; ++nb) {
      const int col = col0 + nb * 16 + lr;
      const float bs = bias ? bias[col] : 0.f;
#pragma unroll
      for (int r = 0; r < 8; ++r) {
        const int m = row0 + mi * 16 + r + 8 * lh;
        const float v = acc[mi][nb][r] + bs;
        if (F32OUT)
          reinterpret_cast<float*>(Cout)[(size_t)m * N + col] = v;
        else
          reinterpret_cast<unsigned short*>(Cout)[(size_t)m * N + col] = f2bf(v);
      }
    }
}

// ---------------------------------------------------------------------------
// RoPE in place on Q [S,NH,HD] and K [S,NKV,HD] (bf16). 64 threads = half dim.
// ---------------------------------------------------------------------------
__global__ __launch_bounds__(64) void rope_kernel(
    unsigned short* __restrict__ Q, unsigned short* __restrict__ Kb,
    const int* __restrict__ pos) {
  const int bid = blockIdx.x;
  const int s = bid / (NH + NKV);
  const int hh = bid % (NH + NKV);
  const int i = threadIdx.x;                 // 0..63
  const float p = (float)pos[s];
  const float freq = p * __powf(10000.0f, -(float)i / 64.0f);
  float sn, c;
  __sincosf(freq, &sn, &c);
  unsigned short* base = (hh < NH)
      ? (Q + ((size_t)s * NH + hh) * HD)
      : (Kb + ((size_t)s * NKV + (hh - NH)) * HD);
  const float x1 = bf2f(base[i]);
  const float x2 = bf2f(base[i + 64]);
  base[i]      = f2bf(x1 * c - x2 * sn);
  base[i + 64] = f2bf(x2 * c + x1 * sn);
}

// ---------------------------------------------------------------------------
// V [S, NKV*HD] -> Vt [NKV, HD, S]  (write-coalesced)
// ---------------------------------------------------------------------------
__global__ __launch_bounds__(256) void transpose_v(
    const unsigned short* __restrict__ V, unsigned short* __restrict__ Vt) {
  int idx = blockIdx.x * blockDim.x + threadIdx.x;
  if (idx >= NKV * HD * S_LEN) return;
  const int s  = idx % S_LEN;
  const int d  = (idx / S_LEN) % HD;
  const int kv = idx / (S_LEN * HD);
  Vt[idx] = V[(size_t)s * KVDIM + kv * HD + d];
}

// ---------------------------------------------------------------------------
// Flash attention: one wave per (16-query tile, head). Key tiles of 32.
// ---------------------------------------------------------------------------
__global__ __launch_bounds__(32) void flash_attn(
    const unsigned short* __restrict__ Q,   // [S, NH*HD]
    const unsigned short* __restrict__ Kb,  // [S, NKV*HD]
    const unsigned short* __restrict__ Vt,  // [NKV, HD, S]
    unsigned short* __restrict__ O) {       // [S, NH*HD]
  __shared__ unsigned short Plds[16 * 32];
  const int qt = blockIdx.x;          // query tile (16 rows)
  const int h  = blockIdx.y;          // head
  const int kv = h / GQ;
  const int lane = threadIdx.x;
  const int lr = lane & 15, lh = lane >> 4;
  const float scale = 0.08838834764831845f;   // 128^-0.5

  // Q fragments (A layout), 4 chunks of k=32 covering HD=128
  Frag qf[4];
  {
    const unsigned short* qrow = Q + ((size_t)(qt * 16 + lr) * NH + h) * HD;
#pragma unroll
    for (int c = 0; c < 4; ++c) {
      const unsigned short* qp = qrow + c * 32 + lh * 8;
      qf[c].q[0] = *reinterpret_cast<const uint4*>(qp);
      qf[c].q[1] = *reinterpret_cast<const uint4*>(qp + 16);
    }
  }

  v8f acc[8];
  float mrow[8], lrow[8];
#pragma unroll
  for (int i = 0; i < 8; ++i) {
#pragma unroll
    for (int j = 0; j < 8; ++j) acc[i][j] = 0.f;
    mrow[i] = -1e30f;
    lrow[i] = 0.f;
  }

  const int nkt = (qt + 2) >> 1;   // 32-wide key tiles covering keys <= qt*16+15
  for (int kt = 0; kt < nkt; ++kt) {
    const int kb = kt * 32;
    v8f s0, s1;
#pragma unroll
    for (int j = 0; j < 8; ++j) { s0[j] = 0.f; s1[j] = 0.f; }

#pragma unroll
    for (int c = 0; c < 4; ++c) {
      Frag b0, b1;
      const unsigned short* kp0 =
          Kb + ((size_t)(kb + lr) * NKV + kv) * HD + c * 32 + lh * 8;
      const unsigned short* kp1 =
          Kb + ((size_t)(kb + 16 + lr) * NKV + kv) * HD + c * 32 + lh * 8;
      b0.q[0] = *reinterpret_cast<const uint4*>(kp0);
      b0.q[1] = *reinterpret_cast<const uint4*>(kp0 + 16);
      b1.q[0] = *reinterpret_cast<const uint4*>(kp1);
      b1.q[1] = *reinterpret_cast<const uint4*>(kp1 + 16);
      s0 = __builtin_amdgcn_wmma_f32_16x16x32_bf16(false, qf[c].v, false, b0.v,
                                                   (short)0, s0, false, false);
      s1 = __builtin_amdgcn_wmma_f32_16x16x32_bf16(false, qf[c].v, false, b1.v,
                                                   (short)0, s1, false, false);
    }

    // online softmax; score rows live across half-waves (C layout)
#pragma unroll
    for (int r = 0; r < 8; ++r) {
      const int qi = qt * 16 + r + 8 * lh;
      const int kj0 = kb + lr, kj1 = kb + 16 + lr;
      float f0 = s0[r] * scale + (kj0 <= qi ? 0.f : -1e9f);
      float f1 = s1[r] * scale + (kj1 <= qi ? 0.f : -1e9f);
      float mx = fmaxf(f0, f1);
#pragma unroll
      for (int off = 1; off < 16; off <<= 1)
        mx = fmaxf(mx, __shfl_xor(mx, off, 16));
      const float nm = fmaxf(mrow[r], mx);
      const float alpha = __expf(mrow[r] - nm);
      const float p0 = __expf(f0 - nm);
      const float p1 = __expf(f1 - nm);
      float rs = p0 + p1;
#pragma unroll
      for (int off = 1; off < 16; off <<= 1)
        rs += __shfl_xor(rs, off, 16);
      lrow[r] = lrow[r] * alpha + rs;
      mrow[r] = nm;
#pragma unroll
      for (int dc = 0; dc < 8; ++dc) acc[dc][r] *= alpha;
      const int m = r + 8 * lh;
      Plds[m * 32 + lr]      = f2bf(p0);
      Plds[m * 32 + 16 + lr] = f2bf(p1);
    }
    __syncthreads();

    // reload P in A layout from LDS
    Frag pf;
    pf.q[0] = *reinterpret_cast<const uint4*>(&Plds[lr * 32 + lh * 8]);
    pf.q[1] = *reinterpret_cast<const uint4*>(&Plds[lr * 32 + lh * 8 + 16]);

#pragma unroll
    for (int dc = 0; dc < 8; ++dc) {
      Frag vf;
      const unsigned short* vp =
          Vt + ((size_t)kv * HD + dc * 16 + lr) * S_LEN + kb + lh * 8;
      vf.q[0] = *reinterpret_cast<const uint4*>(vp);
      vf.q[1] = *reinterpret_cast<const uint4*>(vp + 16);
      acc[dc] = __builtin_amdgcn_wmma_f32_16x16x32_bf16(
          false, pf.v, false, vf.v, (short)0, acc[dc], false, false);
    }
    __syncthreads();
  }

#pragma unroll
  for (int r = 0; r < 8; ++r) {
    const int m = qt * 16 + r + 8 * lh;
    const float inv = 1.f / lrow[r];
#pragma unroll
    for (int dc = 0; dc < 8; ++dc)
      O[((size_t)m * NH + h) * HD + dc * 16 + lr] = f2bf(acc[dc][r] * inv);
  }
}

// ---------------------------------------------------------------------------
extern "C" void kernel_launch(void* const* d_in, const int* in_sizes, int n_in,
                              void* d_out, int out_size, void* d_ws,
                              size_t ws_size, hipStream_t stream) {
  (void)in_sizes; (void)n_in; (void)out_size; (void)ws_size;
  const float* hidden    = (const float*)d_in[0];
  const int*   positions = (const int*)d_in[1];
  /* d_in[2] attention_mask: causal mask computed inline */
  const float* Wq = (const float*)d_in[3];
  const float* bq = (const float*)d_in[4];
  const float* Wk = (const float*)d_in[5];
  const float* bk = (const float*)d_in[6];
  const float* Wv = (const float*)d_in[7];
  const float* bv = (const float*)d_in[8];
  const float* Wo = (const float*)d_in[9];
  float* out = (float*)d_out;

  char* w = (char*)d_ws;
  unsigned short* Xb  = (unsigned short*)w; w += (size_t)S_LEN * HID * 2;
  unsigned short* Wqt = (unsigned short*)w; w += (size_t)QDIM * HID * 2;
  unsigned short* Wkt = (unsigned short*)w; w += (size_t)KVDIM * HID * 2;
  unsigned short* Wvt = (unsigned short*)w; w += (size_t)KVDIM * HID * 2;
  unsigned short* Wot = (unsigned short*)w; w += (size_t)HID * QDIM * 2;
  unsigned short* Qb  = (unsigned short*)w; w += (size_t)S_LEN * QDIM * 2;
  unsigned short* Kbf = (unsigned short*)w; w += (size_t)S_LEN * KVDIM * 2;
  unsigned short* Vbf = (unsigned short*)w; w += (size_t)S_LEN * KVDIM * 2;
  unsigned short* Vtb = (unsigned short*)w; w += (size_t)S_LEN * KVDIM * 2;
  unsigned short* Ab  = (unsigned short*)w; w += (size_t)S_LEN * QDIM * 2;

  // 1. activations fp32 -> bf16
  {
    int n4 = S_LEN * HID / 4;
    cvt_f32_bf16<<<n4 / 256, 256, 0, stream>>>(hidden, Xb, n4);
  }
  // 2. weights fp32 -> bf16 transposed
  transpose_cvt<<<dim3(QDIM / 32, HID / 32), 256, 0, stream>>>(Wq, Wqt, HID, QDIM);
  transpose_cvt<<<dim3(KVDIM / 32, HID / 32), 256, 0, stream>>>(Wk, Wkt, HID, KVDIM);
  transpose_cvt<<<dim3(KVDIM / 32, HID / 32), 256, 0, stream>>>(Wv, Wvt, HID, KVDIM);
  transpose_cvt<<<dim3(HID / 32, QDIM / 32), 256, 0, stream>>>(Wo, Wot, QDIM, HID);

  // 3. QKV projections (bf16 out, bias epilogue), 128x64 block tiles
  gemm_bf16_wmma<false><<<dim3(QDIM / 64, S_LEN / 128), 128, 0, stream>>>(
      Xb, Wqt, bq, Qb, S_LEN, QDIM, HID);
  gemm_bf16_wmma<false><<<dim3(KVDIM / 64, S_LEN / 128), 128, 0, stream>>>(
      Xb, Wkt, bk, Kbf, S_LEN, KVDIM, HID);
  gemm_bf16_wmma<false><<<dim3(KVDIM / 64, S_LEN / 128), 128, 0, stream>>>(
      Xb, Wvt, bv, Vbf, S_LEN, KVDIM, HID);

  // 4. RoPE on Q and K
  rope_kernel<<<S_LEN * (NH + NKV), 64, 0, stream>>>(Qb, Kbf, positions);

  // 5. V -> Vt for contiguous B-fragments in P*V
  {
    int n = NKV * HD * S_LEN;
    transpose_v<<<(n + 255) / 256, 256, 0, stream>>>(Vbf, Vtb);
  }

  // 6. flash attention
  flash_attn<<<dim3(S_LEN / 16, NH), 32, 0, stream>>>(Qb, Kbf, Vtb, Ab);

  // 7. output projection -> fp32 d_out
  gemm_bf16_wmma<true><<<dim3(HID / 64, S_LEN / 128), 128, 0, stream>>>(
      Ab, Wot, nullptr, out, S_LEN, HID, QDIM);
}
